// EfficientGlobalPointer_43190191129200
// MI455X (gfx1250) — compile-verified, matrix operand order checked by reading, not run on hardware
//
#include <hip/hip_runtime.h>
#include <hip/hip_bf16.h>

typedef __attribute__((ext_vector_type(16))) _Float16 v16h;
typedef __attribute__((ext_vector_type(8)))  _Float16 v8h;
typedef __attribute__((ext_vector_type(8)))  float    v8f;

#define B_    16
#define N_    512
#define H_    1024
#define HS_   64      // HEAD_SIZE
#define HEADS_ 12
#define INF_  10000.0f

// ---------------------------------------------------------------------------
// Operand loader for v_wmma_f32_16x16x32_f16.
// CDNA5 16-bit A operand (16x32, M x K): lane L<16 holds row M=L with
// K = {kb..kb+7} in halves [0..7] and K = {kb+16..kb+23} in halves [8..15],
// where kb = chunk base; lanes 16..31 hold the same rows with kb+8.
// B operand mirrors this with N = lane%16, so for C = A*B^T both operands
// load identically from row-major MxK / NxK f16 storage.
// rowp must be 16-byte aligned at rowp+kbase (kbase multiple of 8).
// ---------------------------------------------------------------------------
__device__ inline v16h load_op_h16(const _Float16* __restrict__ rowp, int kbase) {
  v8h lo = *(const v8h*)(rowp + kbase);
  v8h hi = *(const v8h*)(rowp + kbase + 16);
  v16h r;
#pragma unroll
  for (int i = 0; i < 8; ++i) { r[i] = lo[i]; r[8 + i] = hi[i]; }
  return r;
}

// Same layout, but sourced from fp32 memory with on-the-fly cvt to f16.
__device__ inline v16h load_op_f32(const float* __restrict__ rowp, int kbase) {
  v16h r;
#pragma unroll
  for (int i = 0; i < 8; ++i) {
    r[i]     = (_Float16)rowp[kbase + i];
    r[8 + i] = (_Float16)rowp[kbase + 16 + i];
  }
  return r;
}

// ---------------------------------------------------------------------------
// Kernel 0: w1 (1024x128 f32, row-major) -> w1t (128x1024 f16, row-major)
// so stage-1 B operands are contiguous-K loads.
// ---------------------------------------------------------------------------
__global__ void egp_transpose_w1(const float* __restrict__ w1,
                                 _Float16* __restrict__ w1t) {
  int idx = blockIdx.x * blockDim.x + threadIdx.x;   // 0 .. 1024*128-1
  int k = idx >> 7;          // 0..1023
  int n = idx & 127;         // 0..127
  w1t[(size_t)n * H_ + k] = (_Float16)w1[(size_t)k * 128 + n];
}

// ---------------------------------------------------------------------------
// Kernel 1: one wave per 16-row slab of x = inputs @ w1 + b1  (16x128 tile),
// K=1024 via 32 x v_wmma_f32_16x16x32_f16 per column tile (8 column tiles).
// Then RoPE -> qwh/kwh (f16, row-major [B*N][64]) and bias vectors
// biasA[b][h][n] = (x@w2+b2)[n][2h]/2, biasB[b][h][n] = ...[n][2h+1]/2.
// grid = B*32 blocks, blockDim = 32 (single wave -> barriers are free).
// ---------------------------------------------------------------------------
__global__ void egp_stage1(const float* __restrict__ inputs,
                           const _Float16* __restrict__ w1t,
                           const float* __restrict__ b1,
                           const float* __restrict__ w2,
                           const float* __restrict__ b2,
                           _Float16* __restrict__ qwh,
                           _Float16* __restrict__ kwh,
                           float* __restrict__ biasA,
                           float* __restrict__ biasB) {
  __shared__ float xtile[16][128];                    // 8 KB

  const int lane = threadIdx.x;                       // 0..31 (wave32)
  const int b    = blockIdx.x >> 5;
  const int m0   = (blockIdx.x & 31) << 4;            // row tile base in [0,512)
  const int col  = lane & 15;
  const int half = (lane >> 4) & 1;
  const int klo  = half ? 8 : 0;

  const float* arow = inputs + ((size_t)(b * N_ + m0 + col)) * H_;

  v8f acc[8];
#pragma unroll
  for (int j = 0; j < 8; ++j) acc[j] = (v8f)(0.0f);

  for (int kc = 0; kc < 32; ++kc) {                   // K = 1024 in chunks of 32
    const int kb = kc * 32 + klo;
    v16h a = load_op_f32(arow, kb);
#pragma unroll
    for (int j = 0; j < 8; ++j) {
      const _Float16* brow = w1t + ((size_t)(j * 16 + col)) * H_;
      v16h bb = load_op_h16(brow, kb);
      acc[j] = __builtin_amdgcn_wmma_f32_16x16x32_f16(
          false, a, false, bb, (short)0, acc[j], false, false);
    }
  }

  // x tile -> LDS (add b1)
#pragma unroll
  for (int j = 0; j < 8; ++j) {
    float bv = b1[j * 16 + col];
#pragma unroll
    for (int r = 0; r < 8; ++r)
      xtile[r + 8 * half][j * 16 + col] = acc[j][r] + bv;
  }
  __syncthreads();

  // RoPE: qw[d]=x[2d], kw[d]=x[2d+1]; pair (t1,t2)=(q[2e],q[2e+1])=(x[4e],x[4e+2])
  for (int it = lane; it < 16 * 32; it += 32) {
    int m = it >> 5;          // 0..15
    int e = it & 31;          // 0..31
    float freq  = __expf(-(float)(2 * e) * (9.210340371976184f / 64.0f)); // 10000^(-2e/64)
    float ang   = (float)(m0 + m) * freq;
    float s, c;
    __sincosf(ang, &s, &c);
    float x0 = xtile[m][4 * e + 0];
    float x1 = xtile[m][4 * e + 1];
    float x2 = xtile[m][4 * e + 2];
    float x3 = xtile[m][4 * e + 3];
    size_t base = ((size_t)(b * N_ + m0 + m)) * HS_;
    qwh[base + 2 * e]     = (_Float16)(x0 * c - x2 * s);
    qwh[base + 2 * e + 1] = (_Float16)(x0 * s + x2 * c);
    kwh[base + 2 * e]     = (_Float16)(x1 * c - x3 * s);
    kwh[base + 2 * e + 1] = (_Float16)(x1 * s + x3 * c);
  }

  // bias = (x @ w2 + b2) / 2, split even/odd output channels
  for (int it = lane; it < 16 * 24; it += 32) {
    int m = it / 24;
    int o = it % 24;
    float acc2 = b2[o];
    for (int ccol = 0; ccol < 128; ++ccol)
      acc2 += xtile[m][ccol] * w2[ccol * 24 + o];
    acc2 *= 0.5f;
    int h = o >> 1;
    size_t off = ((size_t)(b * HEADS_ + h)) * N_ + (m0 + m);
    if (o & 1) biasB[off] = acc2; else biasA[off] = acc2;
  }
}

// ---------------------------------------------------------------------------
// Kernel 2 (bandwidth-dominant): one wave per 16x16 logits tile.
// logits tile = qw(16x64) * kw(16x64)^T via 2x v_wmma_f32_16x16x32_f16,
// then fan out across 12 heads: + biasA[b,h,n] + biasB[b,h,m] - masks.
// 201 MB of coalesced fp32 stores -> pure HBM-store roofline.
// grid = (32 ntiles, 32 mtiles, 16 batches), blockDim = 32.
// ---------------------------------------------------------------------------
__global__ void egp_stage2(const _Float16* __restrict__ qwh,
                           const _Float16* __restrict__ kwh,
                           const float* __restrict__ biasA,
                           const float* __restrict__ biasB,
                           const int* __restrict__ amask,
                           float* __restrict__ out) {
  const int lane = threadIdx.x;
  const int n0   = blockIdx.x << 4;
  const int m0   = blockIdx.y << 4;
  const int b    = blockIdx.z;
  const int col  = lane & 15;
  const int half = (lane >> 4) & 1;
  const int klo  = half ? 8 : 0;

  const _Float16* arow = qwh + ((size_t)(b * N_ + m0 + col)) * HS_;
  const _Float16* brow = kwh + ((size_t)(b * N_ + n0 + col)) * HS_;

  v8f acc = (v8f)(0.0f);
  {
    v16h a0 = load_op_h16(arow, 0 + klo);
    v16h b0 = load_op_h16(brow, 0 + klo);
    acc = __builtin_amdgcn_wmma_f32_16x16x32_f16(false, a0, false, b0,
                                                 (short)0, acc, false, false);
    v16h a1 = load_op_h16(arow, 32 + klo);
    v16h b1v = load_op_h16(brow, 32 + klo);
    acc = __builtin_amdgcn_wmma_f32_16x16x32_f16(false, a1, false, b1v,
                                                 (short)0, acc, false, false);
  }

  const int n    = n0 + col;                    // column owned by this lane
  const int am_n = amask[b * N_ + n];

  int   mrow[8];
  float pen[8];                                  // INF penalties (pad + tril)
#pragma unroll
  for (int r = 0; r < 8; ++r) {
    mrow[r] = m0 + r + 8 * half;
    int am_m = amask[b * N_ + mrow[r]];
    float p = 0.0f;
    if ((am_m & am_n) == 0) p += INF_;           // pad mask
    if (mrow[r] > n)        p += INF_;           // strict lower triangle
    pen[r] = p;
    acc[r] = acc[r] * 0.125f;                    // 1/sqrt(64)
  }

#pragma unroll
  for (int h = 0; h < HEADS_; ++h) {
    float bA = biasA[((size_t)(b * HEADS_ + h)) * N_ + n];
#pragma unroll
    for (int r = 0; r < 8; ++r) {
      float bBv = biasB[((size_t)(b * HEADS_ + h)) * N_ + mrow[r]];
      float v = acc[r] + bA + bBv - pen[r];
      out[(((size_t)(b * HEADS_ + h)) * N_ + mrow[r]) * N_ + n] = v;
    }
  }
}

// ---------------------------------------------------------------------------
extern "C" void kernel_launch(void* const* d_in, const int* in_sizes, int n_in,
                              void* d_out, int out_size, void* d_ws, size_t ws_size,
                              hipStream_t stream) {
  const float* inputs = (const float*)d_in[0];   // (16,512,1024) f32
  const int*   amask  = (const int*)  d_in[1];   // (16,512) i32
  const float* w1     = (const float*)d_in[2];   // (1024,128)
  const float* b1     = (const float*)d_in[3];   // (128,)
  const float* w2     = (const float*)d_in[4];   // (128,24)
  const float* b2     = (const float*)d_in[5];   // (24,)
  float*       out    = (float*)d_out;           // (16,12,512,512) f32

  char* ws = (char*)d_ws;
  _Float16* qwh   = (_Float16*)(ws);                         // 1 MB
  _Float16* kwh   = (_Float16*)(ws + (1u << 20));            // 1 MB
  float*    biasA = (float*)   (ws + (2u << 20));            // 384 KB
  float*    biasB = (float*)   (ws + (2u << 20) + 393216u);  // 384 KB
  _Float16* w1t   = (_Float16*)(ws + (2u << 20) + 786432u);  // 256 KB

  egp_transpose_w1<<<(H_ * 128) / 256, 256, 0, stream>>>(w1, w1t);

  egp_stage1<<<B_ * (N_ / 16), 32, 0, stream>>>(
      inputs, w1t, b1, w2, b2, qwh, kwh, biasA, biasB);

  dim3 g2(N_ / 16, N_ / 16, B_);
  egp_stage2<<<g2, 32, 0, stream>>>(qwh, kwh, biasA, biasB, amask, out);
}
